// RPN_57226144252474
// MI455X (gfx1250) — matrix operand order repaired and checked by reading, not compiled, not run on hardware
//
#include <hip/hip_runtime.h>
#include <hip/hip_bf16.h>
#include <cmath>

typedef __attribute__((ext_vector_type(16))) _Float16 v16h;
typedef __attribute__((ext_vector_type(8)))  _Float16 v8h;
typedef __attribute__((ext_vector_type(8)))  float    v8f;

struct Anchors { float ax[9], ay[9], aw[9], ah[9]; };

// ---------------------------------------------------------------------------
// Fragment helpers (CDNA5 wave32 WMMA; contiguous k order chosen by this file, same for A and B)
// A (16x32 f16): lane L<16 -> M=L, K[0..15]; lane>=16 -> M=L-16, K[16..31]
// B (32x16 f16): lane L<16 -> N=L, halves = K[0..15]; lane>=16 -> N=L-16, K[16..31]
// ---------------------------------------------------------------------------
__device__ __forceinline__ v16h load_a_frag(const _Float16* Am, int ml, int hi) {
    v8h lo = *(const v8h*)(Am + ml * 32 + hi * 16);
    v8h hh = *(const v8h*)(Am + ml * 32 + hi * 16 + 8);
    v16h a;
#pragma unroll
    for (int i = 0; i < 8; ++i) { a[i] = lo[i]; a[8 + i] = hh[i]; }
    return a;
}
__device__ __forceinline__ v16h load_b_frag(const _Float16* Bt, int n, int hi) {
    return *(const v16h*)(Bt + n * 32 + hi * 16);
}

// ---------------------------------------------------------------------------
// Pack conv weights: (24,512,3,3) f32 -> Wp[t(9)][q(16)][n(32)][c(32)] f16
// ---------------------------------------------------------------------------
__global__ void pack_conv_w(const float* __restrict__ w, _Float16* __restrict__ Wp) {
    int i = blockIdx.x * 256 + threadIdx.x;
    if (i >= 9 * 16 * 32 * 32) return;
    int ci = i & 31;
    int n  = (i >> 5) & 31;
    int q  = (i >> 10) & 15;
    int t  = i >> 14;
    int c  = q * 32 + ci;
    float v = 0.f;
    if (n < 24) v = w[(n * 512 + c) * 9 + t];   // (n,c,r,s), r*3+s = t
    Wp[i] = (_Float16)v;
}

// ---------------------------------------------------------------------------
// Pack head weights: cls_w(18,24) + reg_w(36,24) -> Bht[n(80)][k(32)] f16
// ---------------------------------------------------------------------------
__global__ void pack_head_w(const float* __restrict__ cls_w, const float* __restrict__ reg_w,
                            _Float16* __restrict__ Bht) {
    int i = blockIdx.x * 256 + threadIdx.x;
    if (i >= 80 * 32) return;
    int k = i & 31;
    int n = i >> 5;
    float v = 0.f;
    if (k < 24) {
        if (n < 18)                  v = cls_w[n * 24 + k];
        else if (n >= 32 && n < 68)  v = reg_w[(n - 32) * 24 + k];
    }
    Bht[i] = (_Float16)v;
}

// ---------------------------------------------------------------------------
// conv3x3 + ReLU as implicit GEMM with WMMA, dy-outer / dx-inner with an
// 18-column halo A strip in LDS (re-used by all 3 dx taps -> 3x less x
// traffic, 6 WMMAs per staged tile). No cross-wave sharing: barriers replaced
// by an explicit wave-level s_wait_dscnt. B fragments load straight from the
// packed global panel (one contiguous 32B read per lane, L2-resident).
// Grid: 1024 blocks (b*64+y) x 128 threads (4 waves, one 16-col M tile each).
// ---------------------------------------------------------------------------
__global__ void __launch_bounds__(128) conv3x3_wmma(
    const float* __restrict__ x, const _Float16* __restrict__ Wp,
    const float* __restrict__ conv_b, _Float16* __restrict__ h_out)
{
    __shared__ _Float16 Ash[4][18 * 32];    // per-wave [m(18 halo cols)][c(32)]

    const int tid  = threadIdx.x;
    const int wave = tid >> 5;
    const int lane = tid & 31;
    const int ml   = lane & 15;
    const int hi   = lane >> 4;

    const int by = blockIdx.x;            // b*64 + y
    const int b  = by >> 6;
    const int y  = by & 63;
    const int colbase = wave * 16;
    _Float16* Aw = Ash[wave];

    v8f acc0 = {}; v8f acc1 = {};

    for (int dy = 0; dy < 3; ++dy) {
        const int row = y + dy - 1;
        if (row < 0 || row > 63) continue;                 // uniform
        for (int q = 0; q < 16; ++q) {
            // ---- stage A halo strip: cols colbase-1 .. colbase+16, lane = channel
            const int c = q * 32 + lane;
            const float* xs = x + (((size_t)b * 512 + c) * 64 + row) * 64;
            {
                const float4* xv = (const float4*)(xs + colbase);   // always in-range
#pragma unroll
                for (int k = 0; k < 4; ++k) {
                    float4 f = xv[k];
                    Aw[(4 * k + 1) * 32 + lane] = (_Float16)f.x;
                    Aw[(4 * k + 2) * 32 + lane] = (_Float16)f.y;
                    Aw[(4 * k + 3) * 32 + lane] = (_Float16)f.z;
                    Aw[(4 * k + 4) * 32 + lane] = (_Float16)f.w;
                }
                float vl = (colbase > 0)       ? xs[colbase - 1]  : 0.f;  // wave-uniform cond
                float vr = (colbase + 16 < 64) ? xs[colbase + 16] : 0.f;  // wave-uniform cond
                Aw[0 * 32 + lane]  = (_Float16)vl;
                Aw[17 * 32 + lane] = (_Float16)vr;
            }
            // wave-level LDS visibility (no cross-wave sharing -> no s_barrier)
            asm volatile("s_wait_dscnt 0x0" ::: "memory");

#pragma unroll
            for (int dx = 0; dx < 3; ++dx) {
                const int t = dy * 3 + dx;
                const _Float16* Bg = Wp + (size_t)(t * 16 + q) * 1024;  // [n(32)][c(32)]
                v16h a  = load_a_frag(Aw + dx * 32, ml, hi);
                v16h b0 = load_b_frag(Bg, ml, hi);
                v16h b1 = load_b_frag(Bg, 16 + ml, hi);
                acc0 = __builtin_amdgcn_wmma_f32_16x16x32_f16(false, a, false, b0,
                                                              (short)0, acc0, false, false);
                acc1 = __builtin_amdgcn_wmma_f32_16x16x32_f16(false, a, false, b1,
                                                              (short)0, acc1, false, false);
            }
            // keep this iteration's frag loads ordered before next stage overwrites
            asm volatile("s_wait_dscnt 0x0" ::: "memory");
        }
    }

    // epilogue: bias + ReLU, store h as f16, layout [pos][n(24)]
    const float bias0 = conv_b[ml];
    const float bias1 = (ml < 8) ? conv_b[16 + ml] : 0.f;
#pragma unroll
    for (int r = 0; r < 8; ++r) {
        const int m   = r + hi * 8;
        const size_t pos = (size_t)by * 64 + colbase + m;
        _Float16* dst = h_out + pos * 24;
        float v0 = acc0[r] + bias0; v0 = v0 > 0.f ? v0 : 0.f;
        dst[ml] = (_Float16)v0;
        if (ml < 8) {
            float v1 = acc1[r] + bias1; v1 = v1 > 0.f ? v1 : 0.f;
            dst[16 + ml] = (_Float16)v1;
        }
    }
}

// ---------------------------------------------------------------------------
// Head: 1x1 cls/reg convs via WMMA (K=24 pad 32, N=80 = 5 tiles), pair
// softmax, reference channel-quadruple decode, bb[0::4] scramble, clip,
// batch-0 mask. B panel staged with CDNA5 async global->LDS DMA.
// One wave per 16-position tile: grid = 4096 blocks of 32.
// ---------------------------------------------------------------------------
__global__ void __launch_bounds__(32) head_wmma(
    const _Float16* __restrict__ h_in, const _Float16* __restrict__ Bht,
    const float* __restrict__ cls_b, const float* __restrict__ reg_b,
    float* __restrict__ prob_ws, float* __restrict__ bb_ws,
    float* __restrict__ mask0, Anchors an)
{
    __shared__ _Float16 Ash[16 * 32];
    __shared__ _Float16 Bsh[80 * 32];
    __shared__ float    Lout[16 * 80];

    const int lane = threadIdx.x;
    const int ml = lane & 15;
    const int hi = lane >> 4;

    const int blk = blockIdx.x;     // b*256 + y*4 + xt
    const int b  = blk >> 8;
    const int y  = (blk >> 2) & 63;
    const int xt = blk & 3;
    const size_t posbase = ((size_t)b * 64 + y) * 64 + xt * 16;

    // ---- async global->LDS stage of the 5120B B panel (flat LDS addr: low
    // 32 bits of the shared-aperture flat address ARE the LDS byte offset)
    {
        const unsigned lbase = (unsigned)(uintptr_t)(&Bsh[0]);
        const char*    gbase = (const char*)Bht;
#pragma unroll
        for (int i = 0; i < 10; ++i) {
            unsigned      loff = lbase + (unsigned)((lane + 32 * i) * 16);
            unsigned long long ga = (unsigned long long)(uintptr_t)(gbase + (lane + 32 * i) * 16);
            asm volatile("global_load_async_to_lds_b128 %0, %1, off"
                         :: "v"(loff), "v"(ga) : "memory");
        }
    }
    // stage A: lanes 0..15 each copy one 24-f16 row (48 B) + zero pad
    if (lane < 16) {
        const uint4* s = (const uint4*)(h_in + (posbase + lane) * 24);
        uint4* d = (uint4*)&Ash[lane * 32];
        d[0] = s[0]; d[1] = s[1]; d[2] = s[2];
        uint4 z = {0u, 0u, 0u, 0u};
        *(uint4*)&Ash[lane * 32 + 24] = z;
    }
    asm volatile("s_wait_asynccnt 0x0" ::: "memory");
    asm volatile("s_wait_dscnt 0x0" ::: "memory");

    v16h a = load_a_frag(Ash, ml, hi);
    v8f acc[5] = {};
#pragma unroll
    for (int t = 0; t < 5; ++t) {
        v16h bf = load_b_frag(Bsh, t * 16 + ml, hi);
        acc[t] = __builtin_amdgcn_wmma_f32_16x16x32_f16(false, a, false, bf,
                                                        (short)0, acc[t], false, false);
    }

    // spill tile to LDS with bias
#pragma unroll
    for (int t = 0; t < 5; ++t) {
        const int n = t * 16 + ml;
        float bias = 0.f;
        if (n < 18) bias = cls_b[n];
        else if (n >= 32 && n < 68) bias = reg_b[n - 32];
#pragma unroll
        for (int r = 0; r < 8; ++r)
            Lout[(r + hi * 8) * 80 + n] = acc[t][r] + bias;
    }
    asm volatile("s_wait_dscnt 0x0" ::: "memory");

    // decode: 16 positions x 9 proposals = 144 items over 32 lanes
    for (int w = lane; w < 144; w += 32) {
        const int m = w / 9, g = w % 9;
        const int xpos = xt * 16 + m;
        const float* L = &Lout[m * 80];
        const float p = 1.f / (1.f + __expf(L[2 * g + 1] - L[2 * g]));
        float D[4];
#pragma unroll
        for (int k = 0; k < 4; ++k) {
            const int c = 4 * g + k;
            float v;
            if (c < 9)       { v = an.ax[c]      + an.aw[c]      * L[32 + 4 * c]; }
            else if (c < 18) { int u = c - 9;  v = an.ay[u] + an.ah[u] * L[32 + 4 * u + 1]; }
            else if (c < 27) { int u = c - 18; v = an.aw[u] * __expf(L[32 + 4 * u + 2]); }
            else             { int u = c - 27; v = an.ah[u] * __expf(L[32 + 4 * u + 3]); }
            D[k] = v;
        }
        float x0 = D[0], y0 = D[1];
        float x1 = x0 + D[2] - 1.f;
        float y1 = y0 + D[3] - 1.f;
        x0 = fminf(fmaxf(x0, 0.f), 1023.f);
        y0 = fminf(fmaxf(y0, 0.f), 1023.f);
        x1 = fminf(fmaxf(x1, 0.f), 1023.f);
        y1 = fminf(fmaxf(y1, 0.f), 1023.f);

        const int o = (y * 64 + xpos) * 9 + g;              // original proposal index
        const int j = (o & 3) * 9216 + (o >> 2);            // bb[0::4] scramble
        float* bp = bb_ws + ((size_t)b * 36864 + j) * 4;
        bp[0] = x0; bp[1] = y0; bp[2] = x1; bp[3] = y1;
        prob_ws[(size_t)b * 36864 + o] = p;
        if (b == 0) mask0[o] = (D[2] >= 16.f && D[3] >= 16.f) ? 1.f : 0.f;
    }
}

// ---------------------------------------------------------------------------
// Per-batch top-600 (stable argmax) + sequential NMS. One block per batch.
// ---------------------------------------------------------------------------
__global__ void __launch_bounds__(256) topk_nms(
    const float* __restrict__ prob_ws, const float* __restrict__ mask0,
    const float* __restrict__ bb_ws, float* __restrict__ scratch,
    float* __restrict__ out)
{
    const int b = blockIdx.x, tid = threadIdx.x;
    __shared__ float rv[256]; __shared__ int ri[256];
    __shared__ float tval[600]; __shared__ int tidx[600];
    __shared__ float bx0[600], by0[600], bx1[600], by1[600], barea[600];
    __shared__ int skeep[600];

    float* sc = scratch + (size_t)b * 36864;
    const float* pb = prob_ws + (size_t)b * 36864;
    for (int i = tid; i < 36864; i += 256)
        sc[i] = (mask0[i] != 0.f) ? pb[i] : -1.0f;
    __syncthreads();

    for (int k = 0; k < 600; ++k) {
        float bv = -INFINITY; int bi = 0x7fffffff;
        for (int i = tid; i < 36864; i += 256) {
            float v = sc[i];
            if (v > bv || (v == bv && i < bi)) { bv = v; bi = i; }
        }
        rv[tid] = bv; ri[tid] = bi;
        __syncthreads();
        for (int s = 128; s > 0; s >>= 1) {
            if (tid < s) {
                float ov = rv[tid + s]; int oi = ri[tid + s];
                if (ov > rv[tid] || (ov == rv[tid] && oi < ri[tid])) { rv[tid] = ov; ri[tid] = oi; }
            }
            __syncthreads();
        }
        if (tid == 0) { tval[k] = rv[0]; tidx[k] = ri[0]; sc[ri[0]] = -INFINITY; }
        __syncthreads();
    }

    for (int k = tid; k < 600; k += 256) {
        const float* bp = bb_ws + ((size_t)b * 36864 + tidx[k]) * 4;
        float a0 = bp[0], a1 = bp[1], a2 = bp[2], a3 = bp[3];
        bx0[k] = a0; by0[k] = a1; bx1[k] = a2; by1[k] = a3;
        barea[k] = (a2 - a0 + 1.f) * (a3 - a1 + 1.f);
        skeep[k] = 1;
    }
    __syncthreads();

    for (int i = 0; i < 600; ++i) {
        const int ki = skeep[i];
        const float ix0 = bx0[i], iy0 = by0[i], ix1 = bx1[i], iy1 = by1[i], ia = barea[i];
        if (ki) {
            for (int j = tid; j < 600; j += 256) {
                if (j > i) {
                    float xx0 = fmaxf(ix0, bx0[j]);
                    float yy0 = fmaxf(iy0, by0[j]);
                    float xx1 = fminf(ix1, bx1[j]);
                    float yy1 = fminf(iy1, by1[j]);
                    float w = fmaxf(xx1 - xx0 + 1.f, 0.f);
                    float h = fmaxf(yy1 - yy0 + 1.f, 0.f);
                    float inter = w * h;
                    float iou = inter / (ia + barea[j] - inter);
                    if (iou > 0.7f) skeep[j] = 0;
                }
            }
        }
        __syncthreads();
    }

    float* offs = out;            // (16,600,4)
    float* ps   = out + 38400;    // (16,600)
    float* kp   = out + 48000;    // (16,600)
    for (int k = tid; k < 600; k += 256) {
        float ox = bx0[k], oy = by0[k];
        offs[(b * 600 + k) * 4 + 0] = ox;
        offs[(b * 600 + k) * 4 + 1] = oy;
        offs[(b * 600 + k) * 4 + 2] = bx1[k] - ox + 1.f;
        offs[(b * 600 + k) * 4 + 3] = by1[k] - oy + 1.f;
        ps[b * 600 + k] = tval[k];
        kp[b * 600 + k] = skeep[k] ? 1.f : 0.f;
    }
}

// ---------------------------------------------------------------------------
// Host: anchors (np.round = banker's), ws carve-up, launches
// ---------------------------------------------------------------------------
static void make_anchors_host(Anchors& an) {
    const double ratios[3] = {0.5, 1.0, 2.0};
    const double scales[3] = {8.0, 16.0, 32.0};
    const double base = 256.0;
    const double cc = 7.5, cr = 7.5;
    int idx = 0;
    for (int i = 0; i < 3; ++i) {
        double w = nearbyint(sqrt(base / ratios[i]));
        double h = nearbyint(w * ratios[i]);
        double wm = 0.5 * (w - 1), hm = 0.5 * (h - 1);
        double a0 = cc - wm, a1 = cr - hm, a2 = cc + wm, a3 = cr + hm;
        double awid = a2 - a0 + 1, aht = a3 - a1 + 1;
        double acw = a0 + 0.5 * (awid - 1), ach = a1 + 0.5 * (aht - 1);
        for (int s = 0; s < 3; ++s) {
            double f0 = acw - 0.5 * (awid * scales[s] - 1.0);
            double f1 = ach - 0.5 * (aht  * scales[s] - 1.0);
            double f2 = acw + 0.5 * (awid * scales[s] - 1.0);
            double f3 = ach + 0.5 * (aht  * scales[s] - 1.0);
            double fw = f2 - f0 + 1, fh = f3 - f1 + 1;
            an.ax[idx] = (float)(f0 + 0.5 * (fw - 1));
            an.ay[idx] = (float)(f1 + 0.5 * (fh - 1));
            an.aw[idx] = (float)fw;
            an.ah[idx] = (float)fh;
            ++idx;
        }
    }
}

extern "C" void kernel_launch(void* const* d_in, const int* in_sizes, int n_in,
                              void* d_out, int out_size, void* d_ws, size_t ws_size,
                              hipStream_t stream) {
    (void)in_sizes; (void)n_in; (void)out_size; (void)ws_size;
    const float* x      = (const float*)d_in[0];
    const float* conv_w = (const float*)d_in[1];
    const float* conv_b = (const float*)d_in[2];
    const float* cls_w  = (const float*)d_in[3];
    const float* cls_b  = (const float*)d_in[4];
    const float* reg_w  = (const float*)d_in[5];
    const float* reg_b  = (const float*)d_in[6];
    float* out = (float*)d_out;

    char* ws = (char*)d_ws;
    _Float16* Wp     = (_Float16*)(ws + 0);          //   294912 B
    _Float16* Bht    = (_Float16*)(ws + 294912);     //     5120 B
    _Float16* hbuf   = (_Float16*)(ws + 300032);     //  3145728 B
    float*    probw  = (float*)(ws + 3445760);       //  2359296 B
    float*    mask0  = (float*)(ws + 5805056);       //   147456 B
    float*    bbw    = (float*)(ws + 5952512);       //  9437184 B
    float*    scr    = (float*)(ws + 15389696);      //  2359296 B  (total ~17.7 MB)

    Anchors an;
    make_anchors_host(an);

    pack_conv_w<<<576, 256, 0, stream>>>(conv_w, Wp);
    pack_head_w<<<10, 256, 0, stream>>>(cls_w, reg_w, Bht);
    conv3x3_wmma<<<1024, 128, 0, stream>>>(x, Wp, conv_b, hbuf);
    head_wmma<<<4096, 32, 0, stream>>>(hbuf, Bht, cls_b, reg_b, probw, bbw, mask0, an);
    topk_nms<<<16, 256, 0, stream>>>(probw, mask0, bbw, scr, out);
}
